// RelGINConv_11897059410194
// MI455X (gfx1250) — matrix operand order repaired and checked by reading, not compiled, not run on hardware
//
#include <hip/hip_runtime.h>

#define N_NODES 50000
#define N_EDGES 800000
#define IN_DIM 128
#define EDGE_DIM 32
#define OUT_DIM 128

typedef __attribute__((ext_vector_type(16))) _Float16 v16h;
typedef __attribute__((ext_vector_type(8)))  _Float16 v8h;
typedef __attribute__((ext_vector_type(8)))  float    v8f;

// Workspace layout, units of _Float16:
//   packed B-layout tiles: [otile][ktile][lane(32)][half(16)]
//   value = W[otile*16 + (lane&15)][ktile*32 + (h<8 ? (lane>>4)*8 + h : 16 + (lane>>4)*8 + h-8)]
#define W1_OFF     0        // 8 ot x 4 kt x 512 = 16384
#define W2_OFF     16384
#define WSELF_OFF  32768
#define WEDGE_OFF  49152    // 8 ot x 1 kt x 512 = 4096
#define WPACK_TOTAL 53248

// ---------------- weight packing ----------------
__global__ void pack_weights(const float* __restrict__ W_ne,
                             const float* __restrict__ W_self,
                             const float* __restrict__ W_edge,
                             _Float16* __restrict__ wp) {
    int idx = blockIdx.x * blockDim.x + threadIdx.x;
    if (idx >= WPACK_TOTAL) return;
    int region, base, ktiles;
    if (idx < W2_OFF)         { region = 0; base = idx - W1_OFF;    ktiles = 4; }
    else if (idx < WSELF_OFF) { region = 1; base = idx - W2_OFF;    ktiles = 4; }
    else if (idx < WEDGE_OFF) { region = 2; base = idx - WSELF_OFF; ktiles = 4; }
    else                      { region = 3; base = idx - WEDGE_OFF; ktiles = 1; }
    int h    = base & 15;
    int lane = (base >> 4) & 31;
    int kt   = (base >> 9) % ktiles;
    int ot   = (base >> 9) / ktiles;
    int o = ot * 16 + (lane & 15);
    int k = kt * 32 + (h < 8 ? (lane >> 4) * 8 + h : 16 + (lane >> 4) * 8 + (h - 8));
    float v;
    if (region == 0)      v = W_ne[(size_t)o * 256 + k];          // W1 = W_ne[:, :128]
    else if (region == 1) v = W_ne[(size_t)o * 256 + 128 + k];    // W2 = W_ne[:, 128:]
    else if (region == 2) v = W_self[(size_t)o * 128 + k];
    else                  v = W_edge[(size_t)o * 32 + k];
    wp[idx] = (_Float16)v;
}

__global__ void zero_out_kernel(float* __restrict__ out, int n) {
    int i = blockIdx.x * blockDim.x + threadIdx.x;
    if (i < n) out[i] = 0.0f;
}

// convert 8 consecutive fp32 -> 8 f16
__device__ __forceinline__ void cvt8(const float* __restrict__ p, _Float16* d) {
    float4 a = *(const float4*)p;
    float4 b = *(const float4*)(p + 4);
    d[0] = (_Float16)a.x; d[1] = (_Float16)a.y; d[2] = (_Float16)a.z; d[3] = (_Float16)a.w;
    d[4] = (_Float16)b.x; d[5] = (_Float16)b.y; d[6] = (_Float16)b.z; d[7] = (_Float16)b.w;
}

#define EDGE_WAVES 4

// ---------------- edge kernel: 1 wave == 16 edges ----------------
__global__ __launch_bounds__(32 * EDGE_WAVES)
void edge_kernel(const float* __restrict__ nfeat, const float* __restrict__ etype,
                 const int* __restrict__ src, const int* __restrict__ dst,
                 const float* __restrict__ b_ne, const _Float16* __restrict__ wp,
                 int* __restrict__ outbits) {
    __shared__ _Float16 ehLds[EDGE_WAVES][16 * 128];   // 16 KB: eh tile per wave, row-major f16

    const int lane  = threadIdx.x & 31;
    const int wave  = threadIdx.x >> 5;
    const int tile  = blockIdx.x * EDGE_WAVES + wave;
    const int e0    = tile * 16;
    const int row16 = lane & 15;      // A-layout row / B,C,D-layout column
    const int grp   = lane >> 4;

    // ---- A_et: etype tile (16x32) directly in WMMA A-layout ----
    v16h a_et;
    {
        const float* p = etype + (size_t)(e0 + row16) * EDGE_DIM;
        _Float16 t[16];
        cvt8(p + 8 * grp,      t);
        cvt8(p + 16 + 8 * grp, t + 8);
        #pragma unroll
        for (int i = 0; i < 16; ++i) a_et[i] = t[i];
    }

    // ---- eh = etype @ W_edge^T  (8 WMMAs), stage to LDS as 16x128 f16 row-major ----
    #pragma unroll
    for (int ot = 0; ot < 8; ++ot) {
        v16h b = *(const v16h*)(wp + WEDGE_OFF + ((size_t)ot * 32 + lane) * 16);
        v8f c = {};
        c = __builtin_amdgcn_wmma_f32_16x16x32_f16(false, a_et, false, b,
                                                   (short)0, c, false, false);
        #pragma unroll
        for (int r = 0; r < 8; ++r) {
            int rr = r + 8 * grp;                         // D-layout row
            ehLds[wave][rr * 128 + ot * 16 + row16] = (_Float16)c[r];
        }
    }
    __syncthreads();

    // ---- accumulators seeded with b_ne (broadcast over rows) ----
    v8f acc[8];
    #pragma unroll
    for (int ot = 0; ot < 8; ++ot) {
        float bb = b_ne[ot * 16 + row16];
        #pragma unroll
        for (int r = 0; r < 8; ++r) acc[ot][r] = bb;
    }

    const int s = src[e0 + row16];
    const float* nfp = nfeat + (size_t)s * IN_DIM;

    // ---- main MLP: 4 K-tiles x 8 out-tiles x 2 GEMMs = 64 WMMAs ----
    #pragma unroll
    for (int kt = 0; kt < 4; ++kt) {
        v16h a_nf;
        {
            _Float16 t[16];
            cvt8(nfp + kt * 32 + 8 * grp,      t);
            cvt8(nfp + kt * 32 + 16 + 8 * grp, t + 8);
            #pragma unroll
            for (int i = 0; i < 16; ++i) a_nf[i] = t[i];
        }
        v16h a_eh;
        {
            const _Float16* lp = &ehLds[wave][row16 * 128 + kt * 32];
            v8h lo = *(const v8h*)(lp + 8 * grp);
            v8h hi = *(const v8h*)(lp + 16 + 8 * grp);
            #pragma unroll
            for (int i = 0; i < 8; ++i) { a_eh[i] = lo[i]; a_eh[8 + i] = hi[i]; }
        }
        #pragma unroll
        for (int ot = 0; ot < 8; ++ot) {
            v16h b1 = *(const v16h*)(wp + W1_OFF + ((size_t)(ot * 4 + kt) * 32 + lane) * 16);
            acc[ot] = __builtin_amdgcn_wmma_f32_16x16x32_f16(false, a_nf, false, b1,
                                                             (short)0, acc[ot], false, false);
            v16h b2 = *(const v16h*)(wp + W2_OFF + ((size_t)(ot * 4 + kt) * 32 + lane) * 16);
            acc[ot] = __builtin_amdgcn_wmma_f32_16x16x32_f16(false, a_eh, false, b2,
                                                             (short)0, acc[ot], false, false);
        }
    }

    // ---- ReLU + segment-max scatter (int atomicMax on non-negative float bits) ----
    int drow[8];
    #pragma unroll
    for (int r = 0; r < 8; ++r) drow[r] = dst[e0 + r + 8 * grp];

    #pragma unroll
    for (int ot = 0; ot < 8; ++ot) {
        #pragma unroll
        for (int r = 0; r < 8; ++r) {
            float v = acc[ot][r];
            if (v > 0.0f) {
                atomicMax(outbits + (size_t)drow[r] * OUT_DIM + ot * 16 + row16,
                          __float_as_int(v));
            }
        }
    }
}

// ---------------- node kernel: 1 wave == 16 nodes ----------------
__global__ __launch_bounds__(64)
void node_kernel(const float* __restrict__ nfeat, const float* __restrict__ b_self,
                 const float* __restrict__ eps, const _Float16* __restrict__ wp,
                 float* __restrict__ out) {
    const int lane = threadIdx.x & 31;
    const int wave = threadIdx.x >> 5;
    const int tile = blockIdx.x * 2 + wave;
    if (tile >= N_NODES / 16) return;           // wave-uniform: EXEC stays all-1 for WMMA
    const int n0    = tile * 16;
    const int row16 = lane & 15;
    const int grp   = lane >> 4;

    const float* nfp = nfeat + (size_t)(n0 + row16) * IN_DIM;

    v8f acc[8];
    #pragma unroll
    for (int ot = 0; ot < 8; ++ot) {
        float bb = b_self[ot * 16 + row16];
        #pragma unroll
        for (int r = 0; r < 8; ++r) acc[ot][r] = bb;
    }

    #pragma unroll
    for (int kt = 0; kt < 4; ++kt) {
        v16h a;
        {
            _Float16 t[16];
            cvt8(nfp + kt * 32 + 8 * grp,      t);
            cvt8(nfp + kt * 32 + 16 + 8 * grp, t + 8);
            #pragma unroll
            for (int i = 0; i < 16; ++i) a[i] = t[i];
        }
        #pragma unroll
        for (int ot = 0; ot < 8; ++ot) {
            v16h b = *(const v16h*)(wp + WSELF_OFF + ((size_t)(ot * 4 + kt) * 32 + lane) * 16);
            acc[ot] = __builtin_amdgcn_wmma_f32_16x16x32_f16(false, a, false, b,
                                                             (short)0, acc[ot], false, false);
        }
    }

    const float e1 = 1.0f + eps[0];
    #pragma unroll
    for (int ot = 0; ot < 8; ++ot) {
        #pragma unroll
        for (int r = 0; r < 8; ++r) {
            int rr = r + 8 * grp;
            size_t off = (size_t)(n0 + rr) * OUT_DIM + ot * 16 + row16;
            out[off] = e1 * out[off] + acc[ot][r];   // out currently holds segment-max
        }
    }
}

extern "C" void kernel_launch(void* const* d_in, const int* in_sizes, int n_in,
                              void* d_out, int out_size, void* d_ws, size_t ws_size,
                              hipStream_t stream) {
    const float* nfeat  = (const float*)d_in[0];
    const float* etype  = (const float*)d_in[1];
    const int*   src    = (const int*)d_in[2];
    const int*   dst    = (const int*)d_in[3];
    const float* W_edge = (const float*)d_in[4];
    const float* W_ne   = (const float*)d_in[5];
    const float* b_ne   = (const float*)d_in[6];
    const float* W_self = (const float*)d_in[7];
    const float* b_self = (const float*)d_in[8];
    const float* eps    = (const float*)d_in[9];
    float*       out    = (float*)d_out;
    _Float16*    wp     = (_Float16*)d_ws;

    pack_weights<<<(WPACK_TOTAL + 255) / 256, 256, 0, stream>>>(W_ne, W_self, W_edge, wp);
    zero_out_kernel<<<(N_NODES * OUT_DIM + 255) / 256, 256, 0, stream>>>(out, N_NODES * OUT_DIM);

    const int edge_tiles = N_EDGES / 16;                       // 50000
    edge_kernel<<<edge_tiles / EDGE_WAVES, 32 * EDGE_WAVES, 0, stream>>>(
        nfeat, etype, src, dst, b_ne, wp, (int*)d_out);

    const int node_tiles = N_NODES / 16;                       // 3125
    node_kernel<<<(node_tiles + 1) / 2, 64, 0, stream>>>(nfeat, b_self, eps, wp, out);
}
